// SwinAttention_65257733096103
// MI455X (gfx1250) — matrix operand order repaired and checked by reading, not compiled, not run on hardware
//
#include <hip/hip_runtime.h>
#include <math.h>

typedef __attribute__((ext_vector_type(16))) __bf16 v16bf;
typedef __attribute__((ext_vector_type(8)))  __bf16 v8bf;
typedef __attribute__((ext_vector_type(8)))  float  v8f;
typedef unsigned int uint32x4 __attribute__((ext_vector_type(4)));
typedef int          int32x8  __attribute__((ext_vector_type(8)));
typedef int          int32x4  __attribute__((ext_vector_type(4)));

#define WIN   8
#define TOK   64
#define HEADS 6
#define QKU   32
#define VU    32
#define CH    192
#define UDIM  576
#define NWIN  4096
#define NTAB  (225 * HEADS)             // 1350 floats

// ---- LDS arena layout (bytes) ---------------------------------------------
#define OFF_A    0                      // sA   bf16 [64][200]  = 25600
#define OFF_QKV  25600                  // sQKV bf16 [64][584]  = 74752 (P overlay)
#define OFF_R2   100352                 // region2 overlays:
                                        //   xf32  f32 [64][192] = 49152
                                        //   sQ  bf16 [6][64][40] = 30720
                                        //   sK  bf16 [6][64][40] = 30720 (@+30720)
                                        //   sVt bf16 [6][32][72] = 27648 (@+61440)
                                        //   sOutF f32 [64][192]  = 49152
#define OFF_TAB  192512                 // tabLDS f32 [1350] = 5400
#define SMEM_BYTES 197920

#if __has_builtin(__builtin_amdgcn_tensor_load_to_lds)
#define HAVE_TDM 1
#else
#define HAVE_TDM 0
#endif

#if HAVE_TDM
struct TdmDesc { uint32x4 g0; int32x8 g1; int32x4 g2; int32x4 g3; };

// 3D tile: dim0=192 contiguous f32, dim1=8 (stride 192), dim2=8 (stride 49152)
__device__ __forceinline__ TdmDesc make_tile_desc(unsigned lds_addr,
                                                  unsigned long long gaddr) {
    TdmDesc d;
    d.g0.x = 1u;                                             // count=1, user mode
    d.g0.y = lds_addr;                                       // LDS byte address
    d.g0.z = (unsigned)(gaddr & 0xFFFFFFFFull);              // global_addr[31:0]
    d.g0.w = (unsigned)((gaddr >> 32) & 0x1FFFFFFull)        // global_addr[56:32]
             | (2u << 30);                                   // type=2 (image)
    unsigned w0 = 2u << 16;                                  // data_size=4B
    unsigned w1 = 192u << 16;                                // tensor_dim0 lo16
    unsigned w2 = 256u << 16;                                // tensor_dim1 lo16
    unsigned w3 = 192u << 16;                                // tile_dim0=192
    unsigned w4 = (8u << 16) | 8u;                           // tile_dim2 | tile_dim1
    unsigned w5 = 192u;                                      // dim0_stride[31:0]
    unsigned w6 = 0xC000u << 16;                             // dim1_stride lo16 (49152)
    unsigned w7 = 0u;                                        // dim1_stride[47:16]
    d.g1 = (int32x8){(int)w0,(int)w1,(int)w2,(int)w3,(int)w4,(int)w5,(int)w6,(int)w7};
    d.g2 = (int32x4){256, 0, 0, 0};                          // tensor_dim2=256
    d.g3 = (int32x4){0, 0, 0, 0};
    return d;
}

#if defined(__clang_major__) && __clang_major__ >= 23
#define TDM_LOAD(d)  __builtin_amdgcn_tensor_load_to_lds((d).g0,(d).g1,(d).g2,(d).g3,(int32x8){0,0,0,0,0,0,0,0},0)
#define TDM_STORE(d) __builtin_amdgcn_tensor_store_from_lds((d).g0,(d).g1,(d).g2,(d).g3,(int32x8){0,0,0,0,0,0,0,0},0)
#else
#define TDM_LOAD(d)  __builtin_amdgcn_tensor_load_to_lds((d).g0,(d).g1,(d).g2,(d).g3,0)
#define TDM_STORE(d) __builtin_amdgcn_tensor_store_from_lds((d).g0,(d).g1,(d).g2,(d).g3,0)
#endif
#endif // HAVE_TDM

// A-fragment (16x32 bf16): lane L -> M=L%16, g=L/16; elem e -> K=16g+e (contiguous)
__device__ __forceinline__ v16bf load_afrag(const __bf16* row, int kbase, int g) {
    const v8bf lo = *reinterpret_cast<const v8bf*>(row + kbase + 16 * g);
    const v8bf hi = *reinterpret_cast<const v8bf*>(row + kbase + 16 * g + 8);
    return __builtin_shufflevector(lo, hi, 0,1,2,3,4,5,6,7,8,9,10,11,12,13,14,15);
}
// B-fragment (32x16 bf16): lane L -> N=L%16, g=L/16; elem e -> K=16g+e (contiguous)
__device__ __forceinline__ v16bf load_bfrag(const __bf16* p) {
    const v8bf lo = *reinterpret_cast<const v8bf*>(p);
    const v8bf hi = *reinterpret_cast<const v8bf*>(p + 8);
    return __builtin_shufflevector(lo, hi, 0,1,2,3,4,5,6,7,8,9,10,11,12,13,14,15);
}

// ---------------------------------------------------------------------------
// Kernel 0: CPB MLP -> tabOut[225][6]
// ---------------------------------------------------------------------------
__global__ void swin_cpb_kernel(const float* __restrict__ w1,
                                const float* __restrict__ b1,
                                const float* __restrict__ w2,
                                float* __restrict__ tabOut) {
    int i = threadIdx.x;
    if (i >= 225) return;
    float c0 = (float)(i / 15 - 7) * (8.0f / 7.0f);
    float c1 = (float)(i % 15 - 7) * (8.0f / 7.0f);
    const float inv_log8 = 1.0f / logf(8.0f);
    float t0 = copysignf(log1pf(fabsf(c0)) * inv_log8, c0);
    float t1 = copysignf(log1pf(fabsf(c1)) * inv_log8, c1);
    float o[HEADS] = {0.f, 0.f, 0.f, 0.f, 0.f, 0.f};
    for (int j = 0; j < 512; ++j) {
        float hid = fmaxf(t0 * w1[j] + t1 * w1[512 + j] + b1[j], 0.0f);
        #pragma unroll
        for (int h = 0; h < HEADS; ++h) o[h] += hid * w2[j * HEADS + h];
    }
    #pragma unroll
    for (int h = 0; h < HEADS; ++h)
        tabOut[i * HEADS + h] = 16.0f / (1.0f + __expf(-o[h]));
}

// ---------------------------------------------------------------------------
// Kernel 0b: transpose + bf16-convert weights, fuse qkv bias, into d_ws
//   bqkvT[d][k] = qkv_w[k][d]   (576 x 192)
//   bprojT[d][k] = proj_w[k][d] (192 x 192)
//   biasf[d]    = d<192 ? q_bias[d] : d<384 ? 0 : v_bias[d-384]
// ---------------------------------------------------------------------------
__global__ void swin_wprep_kernel(const float* __restrict__ qkv_w,
                                  const float* __restrict__ proj_w,
                                  const float* __restrict__ q_bias,
                                  const float* __restrict__ v_bias,
                                  unsigned short* __restrict__ bqkvT_u,
                                  unsigned short* __restrict__ bprojT_u,
                                  float* __restrict__ biasf) {
    __bf16* bqkvT  = reinterpret_cast<__bf16*>(bqkvT_u);
    __bf16* bprojT = reinterpret_cast<__bf16*>(bprojT_u);
    int i = blockIdx.x * 256 + threadIdx.x;
    if (i < UDIM * CH) {
        int d = i / CH, k = i % CH;
        bqkvT[i] = (__bf16)qkv_w[k * UDIM + d];
    }
    if (i < CH * CH) {
        int d = i / CH, k = i % CH;
        bprojT[i] = (__bf16)proj_w[k * CH + d];
    }
    if (i < UDIM)
        biasf[i] = (i < 192) ? q_bias[i] : ((i < 384) ? 0.0f : v_bias[i - 384]);
}

// ---------------------------------------------------------------------------
// Fused per-window kernel (256 thr = 8 waves per 8x8 window)
// ---------------------------------------------------------------------------
template <bool BF16W>
__global__ void __launch_bounds__(256)
swin_window_kernel(const float* __restrict__ x,
                   const float* __restrict__ qkv_w,
                   const unsigned short* __restrict__ bqkvT_u,
                   const float* __restrict__ q_bias,
                   const float* __restrict__ v_bias,
                   const float* __restrict__ biasf,
                   const float* __restrict__ scale,
                   const float* __restrict__ tabOut,
                   const float* __restrict__ proj_w,
                   const unsigned short* __restrict__ bprojT_u,
                   const float* __restrict__ proj_b,
                   float* __restrict__ out) {
    __shared__ __attribute__((aligned(16))) char smem[SMEM_BYTES];
    __bf16 (*sA)[200]   = reinterpret_cast<__bf16(*)[200]>(smem + OFF_A);
    __bf16 (*sQKV)[584] = reinterpret_cast<__bf16(*)[584]>(smem + OFF_QKV);
    typedef __bf16 QArr[TOK][40];
    typedef __bf16 VArr[QKU][72];
    QArr*  sQ           = reinterpret_cast<QArr*>(smem + OFF_R2);
    QArr*  sK           = reinterpret_cast<QArr*>(smem + OFF_R2 + 30720);
    VArr*  sV           = reinterpret_cast<VArr*>(smem + OFF_R2 + 61440);   // transposed V
    float (*xf32)[CH]   = reinterpret_cast<float(*)[CH]>(smem + OFF_R2);
    float (*sOutF)[CH]  = reinterpret_cast<float(*)[CH]>(smem + OFF_R2);
    float* tabLDS       = reinterpret_cast<float*>(smem + OFF_TAB);

    const __bf16* bqkvT  = reinterpret_cast<const __bf16*>(bqkvT_u);
    const __bf16* bprojT = reinterpret_cast<const __bf16*>(bprojT_u);

    const int n  = blockIdx.x;
    const int b  = n >> 10;
    const int h0 = ((n & 1023) >> 5) * WIN;
    const int w0 = (n & 31) * WIN;
    const size_t tile_elem = (((size_t)b * 256 + h0) * 256 + w0) * CH;

    const int tid    = threadIdx.x;
    const int wv     = tid >> 5;
    const int lane   = tid & 31;
    const int lane16 = lane & 15;
    const int g      = lane >> 4;

    // ---------------- Phase 0: TDM x tile -> LDS; stage CPB table ----------
#if HAVE_TDM
    if (wv == 0) {
        TdmDesc dl = make_tile_desc(
            (unsigned)(uintptr_t)&xf32[0][0],
            (unsigned long long)(uintptr_t)x + tile_elem * 4ull);
        TDM_LOAD(dl);
    }
#endif
    for (int i = tid; i < NTAB; i += 256) tabLDS[i] = tabOut[i];
#if HAVE_TDM
    if (wv == 0) __builtin_amdgcn_s_wait_tensorcnt(0);
    __syncthreads();
    for (int f = tid; f < (TOK * CH) / 4; f += 256) {
        int t  = f / 48;
        int cq = (f % 48) * 4;
        const float4 v = *reinterpret_cast<const float4*>(&xf32[t][cq]);
        sA[t][cq + 0] = (__bf16)v.x;
        sA[t][cq + 1] = (__bf16)v.y;
        sA[t][cq + 2] = (__bf16)v.z;
        sA[t][cq + 3] = (__bf16)v.w;
    }
#else
    for (int f = tid; f < (TOK * CH) / 4; f += 256) {
        int t  = f / 48;
        int cq = (f % 48) * 4;
        int py = h0 + (t >> 3), px = w0 + (t & 7);
        const float4 v = *reinterpret_cast<const float4*>(
            &x[(((size_t)b * 256 + py) * 256 + px) * CH + cq]);
        sA[t][cq + 0] = (__bf16)v.x;
        sA[t][cq + 1] = (__bf16)v.y;
        sA[t][cq + 2] = (__bf16)v.z;
        sA[t][cq + 3] = (__bf16)v.w;
    }
#endif
    __syncthreads();

    // ---------------- Phase 2: qkv GEMM (64x576x192) -----------------------
    {
        const int mt = wv >> 1;
        const int ng = wv & 1;
        const int m  = mt * 16 + lane16;
        v16bf afr[6];
        #pragma unroll
        for (int ks = 0; ks < 6; ++ks) afr[ks] = load_afrag(&sA[m][0], ks * 32, g);
        for (int i = 0; i < 18; ++i) {
            int nt   = ng * 18 + i;
            int ncol = nt * 16 + lane16;
            v8f acc = {};
            #pragma unroll
            for (int ks = 0; ks < 6; ++ks) {
                v16bf bfr;
                if constexpr (BF16W) {
                    bfr = load_bfrag(&bqkvT[ncol * CH + ks * 32 + 16 * g]);
                } else {
                    #pragma unroll
                    for (int e = 0; e < 16; ++e)
                        bfr[e] = (__bf16)qkv_w[(ks * 32 + 16 * g + e) * UDIM + ncol];
                }
                acc = __builtin_amdgcn_wmma_f32_16x16x32_bf16(
                          false, afr[ks], false, bfr, (short)0, acc, false, false);
            }
            float bias;
            if constexpr (BF16W) {
                bias = biasf[ncol];
            } else {
                bias = (ncol < 192) ? q_bias[ncol]
                      : ((ncol < 384) ? 0.0f : v_bias[ncol - 384]);
            }
            #pragma unroll
            for (int r = 0; r < 8; ++r)
                sQKV[mt * 16 + 8 * g + r][ncol] = (__bf16)(acc[r] + bias);
        }
    }
    __syncthreads();

    // ---------------- Phase 3: l2norm q,k; temperature; V transposed -------
    for (int p = tid; p < TOK * HEADS; p += 256) {
        int h = p % HEADS;
        int t = p / HEADS;
        float ls = __expf(fminf(scale[h], 4.60517019f));
        float buf[QKU];
        float ss = 0.0f;
        #pragma unroll
        for (int j = 0; j < QKU; ++j) {
            float q = (float)sQKV[t][h * 96 + j];
            buf[j] = q; ss += q * q;
        }
        float rinv = rsqrtf(fmaxf(ss, 1.55e-05f)) * ls;
        #pragma unroll
        for (int j = 0; j < QKU; ++j) sQ[h][t][j] = (__bf16)(buf[j] * rinv);
        ss = 0.0f;
        #pragma unroll
        for (int j = 0; j < QKU; ++j) {
            float k = (float)sQKV[t][h * 96 + 32 + j];
            buf[j] = k; ss += k * k;
        }
        rinv = rsqrtf(fmaxf(ss, 1.55e-05f));
        #pragma unroll
        for (int j = 0; j < QKU; ++j) sK[h][t][j] = (__bf16)(buf[j] * rinv);
        #pragma unroll
        for (int j = 0; j < VU; ++j) sV[h][j][t] = sQKV[t][h * 96 + 64 + j];
    }
    __syncthreads();

    // ---------------- Phase 4: attention, one wave per head ----------------
    if (wv < HEADS) {
        const int h = wv;
        __bf16* sP = &sQKV[0][0] + (size_t)h * TOK * 72;   // [64][72]

        v16bf qA[4];
        #pragma unroll
        for (int mt = 0; mt < 4; ++mt)
            qA[mt] = load_afrag(&sQ[h][mt * 16 + lane16][0], 0, g);

        v8f S[4][4];
        #pragma unroll
        for (int nt = 0; nt < 4; ++nt) {
            v16bf bfr = load_bfrag(&sK[h][nt * 16 + lane16][16 * g]);
            #pragma unroll
            for (int mt = 0; mt < 4; ++mt) {
                v8f z = {};
                S[mt][nt] = __builtin_amdgcn_wmma_f32_16x16x32_bf16(
                                false, qA[mt], false, bfr, (short)0, z, false, false);
            }
        }

        #pragma unroll
        for (int mt = 0; mt < 4; ++mt) {
            #pragma unroll
            for (int nt = 0; nt < 4; ++nt) {
                int nk = nt * 16 + lane16;
                int kr = nk >> 3, kc = nk & 7;
                #pragma unroll
                for (int r = 0; r < 8; ++r) {
                    int mq = mt * 16 + 8 * g + r;
                    int idx = ((mq >> 3) - kr + 7) * 15 + ((mq & 7) - kc + 7);
                    S[mt][nt][r] += tabLDS[idx * HEADS + h];
                }
            }
            #pragma unroll
            for (int r = 0; r < 8; ++r) {
                float mx = fmaxf(fmaxf(S[mt][0][r], S[mt][1][r]),
                                 fmaxf(S[mt][2][r], S[mt][3][r]));
                #pragma unroll
                for (int off = 1; off < 16; off <<= 1)
                    mx = fmaxf(mx, __shfl_xor(mx, off));
                float sum = 0.0f;
                #pragma unroll
                for (int nt = 0; nt < 4; ++nt) {
                    float pv = __expf(S[mt][nt][r] - mx);
                    S[mt][nt][r] = pv; sum += pv;
                }
                #pragma unroll
                for (int off = 1; off < 16; off <<= 1)
                    sum += __shfl_xor(sum, off);
                float inv = 1.0f / sum;
                #pragma unroll
                for (int nt = 0; nt < 4; ++nt) S[mt][nt][r] *= inv;
            }
            #pragma unroll
            for (int nt = 0; nt < 4; ++nt)
                #pragma unroll
                for (int r = 0; r < 8; ++r)
                    sP[(mt * 16 + 8 * g + r) * 72 + nt * 16 + lane16] =
                        (__bf16)S[mt][nt][r];
        }

        v8f O[4][2] = {};
        #pragma unroll
        for (int ks = 0; ks < 2; ++ks) {
            v16bf pA[4];
            #pragma unroll
            for (int mt = 0; mt < 4; ++mt)
                pA[mt] = load_afrag(sP + (mt * 16 + lane16) * 72, ks * 32, g);
            #pragma unroll
            for (int ntv = 0; ntv < 2; ++ntv) {
                v16bf bfr = load_bfrag(&sV[h][ntv * 16 + lane16][ks * 32 + 16 * g]);
                #pragma unroll
                for (int mt = 0; mt < 4; ++mt)
                    O[mt][ntv] = __builtin_amdgcn_wmma_f32_16x16x32_bf16(
                                     false, pA[mt], false, bfr, (short)0,
                                     O[mt][ntv], false, false);
            }
        }
        #pragma unroll
        for (int mt = 0; mt < 4; ++mt)
            #pragma unroll
            for (int ntv = 0; ntv < 2; ++ntv)
                #pragma unroll
                for (int r = 0; r < 8; ++r)
                    sA[mt * 16 + 8 * g + r][h * 32 + ntv * 16 + lane16] =
                        (__bf16)O[mt][ntv][r];
    }
    __syncthreads();

    // ---------------- Phase 5: proj GEMM (64x192x192) + bias ---------------
    {
        const int mt   = wv >> 1;
        const int half = wv & 1;
        const int m    = mt * 16 + lane16;
        v16bf afr[6];
        #pragma unroll
        for (int ks = 0; ks < 6; ++ks) afr[ks] = load_afrag(&sA[m][0], ks * 32, g);
        #pragma unroll
        for (int i = 0; i < 6; ++i) {
            int ncol = (half * 6 + i) * 16 + lane16;
            v8f acc = {};
            #pragma unroll
            for (int ks = 0; ks < 6; ++ks) {
                v16bf bfr;
                if constexpr (BF16W) {
                    bfr = load_bfrag(&bprojT[ncol * CH + ks * 32 + 16 * g]);
                } else {
                    #pragma unroll
                    for (int e = 0; e < 16; ++e)
                        bfr[e] = (__bf16)proj_w[(ks * 32 + 16 * g + e) * CH + ncol];
                }
                acc = __builtin_amdgcn_wmma_f32_16x16x32_bf16(
                          false, afr[ks], false, bfr, (short)0, acc, false, false);
            }
            float pb = proj_b[ncol];
            #pragma unroll
            for (int r = 0; r < 8; ++r) {
                int row = mt * 16 + 8 * g + r;
#if HAVE_TDM
                sOutF[row][ncol] = acc[r] + pb;
#else
                int py = h0 + (row >> 3), px = w0 + (row & 7);
                out[(((size_t)b * 256 + py) * 256 + px) * CH + ncol] = acc[r] + pb;
#endif
            }
        }
    }
#if HAVE_TDM
    __syncthreads();
    if (wv == 0) {
        TdmDesc dso = make_tile_desc(
            (unsigned)(uintptr_t)&sOutF[0][0],
            (unsigned long long)(uintptr_t)out + tile_elem * 4ull);
        TDM_STORE(dso);
        __builtin_amdgcn_s_wait_tensorcnt(0);
    }
#endif
}

// ---------------------------------------------------------------------------
extern "C" void kernel_launch(void* const* d_in, const int* in_sizes, int n_in,
                              void* d_out, int out_size, void* d_ws, size_t ws_size,
                              hipStream_t stream) {
    const float* x      = (const float*)d_in[0];
    const float* qkv_w  = (const float*)d_in[1];
    const float* q_bias = (const float*)d_in[2];
    const float* v_bias = (const float*)d_in[3];
    const float* scale  = (const float*)d_in[4];
    const float* cpb_w1 = (const float*)d_in[5];
    const float* cpb_b1 = (const float*)d_in[6];
    const float* cpb_w2 = (const float*)d_in[7];
    const float* proj_w = (const float*)d_in[8];
    const float* proj_b = (const float*)d_in[9];
    float*       out    = (float*)d_out;

    float*          tabOut = (float*)d_ws;                               // 225*6 f32
    unsigned short* bqkvT  = (unsigned short*)((char*)d_ws + 8192);      // 576*192 bf16
    unsigned short* bprojT = (unsigned short*)((char*)d_ws + 8192 + UDIM * CH * 2);
    float*          biasf  = (float*)((char*)d_ws + 8192 + UDIM * CH * 2 + CH * CH * 2);
    const size_t ws_need = 8192 + (size_t)UDIM * CH * 2 + (size_t)CH * CH * 2
                         + (size_t)UDIM * 4;

    swin_cpb_kernel<<<1, 256, 0, stream>>>(cpb_w1, cpb_b1, cpb_w2, tabOut);

    if (ws_size >= ws_need) {
        swin_wprep_kernel<<<(UDIM * CH + 255) / 256, 256, 0, stream>>>(
            qkv_w, proj_w, q_bias, v_bias, bqkvT, bprojT, biasf);
        swin_window_kernel<true><<<NWIN, 256, 0, stream>>>(
            x, qkv_w, bqkvT, q_bias, v_bias, biasf, scale, tabOut,
            proj_w, bprojT, proj_b, out);
    } else {
        swin_window_kernel<false><<<NWIN, 256, 0, stream>>>(
            x, qkv_w, bqkvT, q_bias, v_bias, biasf, scale, tabOut,
            proj_w, bprojT, proj_b, out);
    }
}
